// RasterTriangle_46566035423850
// MI455X (gfx1250) — matrix operand order repaired and checked by reading, not compiled, not run on hardware
//
#include <hip/hip_runtime.h>

typedef __attribute__((ext_vector_type(16))) _Float16 v16h;
typedef __attribute__((ext_vector_type(8)))  _Float16 v8h;
typedef __attribute__((ext_vector_type(8)))  float    v8f;

union Frag16 {
    v16h v;
    v8h  h[2];
};

#define TILE    128
#define RSTRIDE 72   // halves per LDS row: 64 data + 8 pad (bank spread)

__global__ __launch_bounds__(256)
void hilbert_rank_wmma_kernel(const float* __restrict__ in00,
                              const float* __restrict__ in01,
                              float* __restrict__ out)
{
    // 1/SIDE = 512 / (pi/2).  Pre-scaling both factors by 1/SIDE makes the
    // WMMA accumulator equal to the final (already /PIXEL_AREA) result.
    const float invSide = (float)(1024.0 / 3.14159265358979323846);

    __shared__ float xs[64];
    __shared__ float ys[64];
    __shared__ _Float16 horL [TILE * RSTRIDE];   // B factor: hor[q][k]
    __shared__ _Float16 vertL[TILE * RSTRIDE];   // A factor: vert[p][k]

    const int tid   = threadIdx.x;
    const int qBase = blockIdx.x * TILE;   // output column tile (hor rows)
    const int pBase = blockIdx.y * TILE;   // output row tile    (vert rows)
    const int bz    = blockIdx.z;
    const int batch = bz >> 1;
    const int plane = bz & 1;              // 0 = hilbert-function, 1 = rank-invariant

    // ---- stage scaled interval endpoints (K = 64, ri zero-padded) ----
    if (tid < 64) {
        const int k = tid;
        float xv, yv;
        if (k < 32) {
            xv = in00[batch * 64 + 2 * k]     * invSide;   // a0 / SIDE
            yv = in00[batch * 64 + 2 * k + 1] * invSide;   // b0 / SIDE
        } else if (plane == 0) {
            xv = in01[batch * 64 + 2 * (k - 32) + 1] * invSide - 512.0f; // (b1-STRIP)/SIDE
            yv = in01[batch * 64 + 2 * (k - 32)]     * invSide;          // a1 / SIDE
        } else {
            xv =  1.0e9f;   // sentinels: force hor = vert = 0 for padded K
            yv = -1.0e9f;
        }
        xs[k] = xv;
        ys[k] = yv;
    }
    __syncthreads();

    // ---- build the f16 factor tiles in LDS (two halves per store pair) ----
    for (int idx = tid; idx < TILE * 32; idx += 256) {
        const int row = idx >> 5;
        const int k   = (idx & 31) * 2;
        const float qf = (float)(qBase + row);
        const float pf = (float)(pBase + row);

        float h0, h1, v0, v1;
        if (plane == 0) {
            // hor  = relu(min(SIDE, PX+SIDE-x, y-PX)) / SIDE
            // vert = relu(min(SIDE, PY+SIDE-y, x+STRIP-PY)) / SIDE
            h0 = fminf(1.0f, fminf(qf + 1.0f - xs[k],     ys[k]     - qf));
            h1 = fminf(1.0f, fminf(qf + 1.0f - xs[k + 1], ys[k + 1] - qf));
            v0 = fminf(1.0f, fminf(512.0f - pf - ys[k],     xs[k]     + 1.0f + pf));
            v1 = fminf(1.0f, fminf(512.0f - pf - ys[k + 1], xs[k + 1] + 1.0f + pf));
        } else {
            // hor  = relu(min(SIDE, PX+SIDE-a0)) / SIDE
            // vert = relu(min(SIDE, b0-PY)) / SIDE
            h0 = fminf(1.0f, qf + 1.0f - xs[k]);
            h1 = fminf(1.0f, qf + 1.0f - xs[k + 1]);
            v0 = fminf(1.0f, ys[k]     - 511.0f + pf);
            v1 = fminf(1.0f, ys[k + 1] - 511.0f + pf);
        }
        h0 = fmaxf(h0, 0.0f); h1 = fmaxf(h1, 0.0f);
        v0 = fmaxf(v0, 0.0f); v1 = fmaxf(v1, 0.0f);

        horL [row * RSTRIDE + k]     = (_Float16)h0;
        horL [row * RSTRIDE + k + 1] = (_Float16)h1;
        vertL[row * RSTRIDE + k]     = (_Float16)v0;
        vertL[row * RSTRIDE + k + 1] = (_Float16)v1;
    }
    __syncthreads();

    // ---- WMMA contraction: D(16x16) = vert(16x64) x hor^T(64x16) ----
    const int wave = tid >> 5;       // 8 waves, one 16-row band each
    const int lane = tid & 31;
    const int lm   = lane & 15;
    const int hi   = lane >> 4;
    const int m0r  = wave * 16;

    // A fragment layout (16-bit, 16x32): lane m  -> K 0-15 (+kc)
    //                                    lane m+16 -> K 16-31 (+kc)
    const _Float16* vrow = &vertL[(m0r + lm) * RSTRIDE];
    Frag16 aLo, aHi;
    aLo.h[0] = *(const v8h*)(vrow +      16 * hi);
    aLo.h[1] = *(const v8h*)(vrow +      16 * hi + 8);
    aHi.h[0] = *(const v8h*)(vrow + 32 + 16 * hi);
    aHi.h[1] = *(const v8h*)(vrow + 32 + 16 * hi + 8);

    float* outPlane = out + (((size_t)batch * 2 + plane) * 512 + pBase) * 512 + qBase;
    const int rowBase = m0r + 8 * hi;   // C/D layout: VGPR r -> row r (+8 for hi half)

    #pragma unroll
    for (int nt = 0; nt < 8; ++nt) {
        const int n0r = nt * 16;
        // B fragment layout (32x16): lane n -> col n, K 0-15; lane n+16 -> K 16-31
        const _Float16* hrow = &horL[(n0r + lm) * RSTRIDE + 16 * hi];
        Frag16 bLo, bHi;
        bLo.h[0] = *(const v8h*)(hrow);
        bLo.h[1] = *(const v8h*)(hrow + 8);
        bHi.h[0] = *(const v8h*)(hrow + 32);
        bHi.h[1] = *(const v8h*)(hrow + 40);

        v8f c = {};
        c = __builtin_amdgcn_wmma_f32_16x16x32_f16(false, aLo.v, false, bLo.v,
                                                   (short)0, c, false, false);
        c = __builtin_amdgcn_wmma_f32_16x16x32_f16(false, aHi.v, false, bHi.v,
                                                   (short)0, c, false, false);

        const int col = n0r + lm;
        #pragma unroll
        for (int r = 0; r < 8; ++r) {
            // Output (512 MiB) exceeds the 192 MB L2: stream with NT hint.
            __builtin_nontemporal_store(c[r], &outPlane[(size_t)(rowBase + r) * 512 + col]);
        }
    }
}

extern "C" void kernel_launch(void* const* d_in, const int* in_sizes, int n_in,
                              void* d_out, int out_size, void* d_ws, size_t ws_size,
                              hipStream_t stream) {
    const float* in00 = (const float*)d_in[0];   // intervals00: (256,32,2) f32
    const float* in01 = (const float*)d_in[1];   // intervals01: (256,32,2) f32
    float* out = (float*)d_out;                  // (256,2,512,512) f32

    dim3 grid(4, 4, 512);   // (qTile, pTile, batch*2+plane)
    dim3 block(256);
    hipLaunchKernelGGL(hilbert_rank_wmma_kernel, grid, block, 0, stream,
                       in00, in01, out);
}
